// attention_action_decoder_18665927868456
// MI455X (gfx1250) — matrix operand order repaired and checked by reading, not compiled, not run on hardware
//
#include <hip/hip_runtime.h>

// ---------------------------------------------------------------------------
// Types for CDNA5 WMMA (wave32): v_wmma_f32_16x16x32_bf16
// ---------------------------------------------------------------------------
typedef __attribute__((ext_vector_type(16))) __bf16 v16bf;
typedef __attribute__((ext_vector_type(8)))  float  v8f;
typedef __attribute__((ext_vector_type(4)))  float  v4f_t;

union FragU { v16bf v; v4f_t q[2]; };

#define B_   128
#define S_   128
#define T_   64
#define H_   1024   // HIDDEN == INPUT_SIZE
#define E_   256    // EMBED == ENV_DIM
#define WD_  1536   // W_D_DIM
#define G_   4096   // 4*HIDDEN
#define AS_  1000   // ACTION_SIZE
#define ASP_ 1024   // padded action size

// ---------------------------------------------------------------------------
// Fragment loaders (bf16), A and B both in the same contiguous K order:
//  A (16x32, MxK): lane L holds row M=L&15;  K = (L>>4)*16 + [0..15] (contiguous)
//  B (32x16, KxN): lane L holds col N=L&15;  K = (L>>4)*16 + [0..15] (contiguous)
// Loads take a single base pointer plus a 32-bit per-lane element offset so
// the backend can use saddr + voffset addressing (minimal VGPR addr pressure).
// ---------------------------------------------------------------------------
__device__ __forceinline__ v16bf load_afrag(const __bf16* base, unsigned off) {
  const __bf16* p = base + off;
  FragU u;
  u.q[0] = *(const v4f_t*)p;        // K chunk 0..7   (+khalf*16)
  u.q[1] = *(const v4f_t*)(p + 8);  // K chunk 8..15  (+khalf*16)
  return u.v;
}

__device__ __forceinline__ v16bf load_bfrag(const __bf16* base, unsigned off) {
  const __bf16* p = base + off;
  FragU u;
  u.q[0] = *(const v4f_t*)p;        // 16 contiguous K values
  u.q[1] = *(const v4f_t*)(p + 8);
  return u.v;
}

// ---------------------------------------------------------------------------
// Double-buffered K-loop: prefetch k+32 fragments into the alternate register
// buffer before the WMMAs on the current one (no WAR hazard -> loads overlap
// matrix ops; compiler can use partial s_wait_loadcnt).  K must be a multiple
// of 64 (all our K: 1024, 1536).  Buffer index is compile-time constant.
// ---------------------------------------------------------------------------
template<int NT, int MT>
__device__ __forceinline__ void gemm_loop(const __bf16* __restrict__ A, int lda,
                                          const __bf16* __restrict__ W, int ldw,
                                          int K, int m0, int n0, int lane,
                                          v8f (&acc)[NT][MT]) {
  const int arow = lane & 15;
  const int khalf = lane >> 4;

  // Per-lane 32-bit element offsets (buffers are < 2^31 elements).
  unsigned aoff[MT], boff[NT];
#pragma unroll
  for (int i = 0; i < MT; i++)
    aoff[i] = (unsigned)(m0 + i * 16 + arow) * (unsigned)lda + (khalf << 4);
#pragma unroll
  for (int j = 0; j < NT; j++)
    boff[j] = (unsigned)(n0 + j * 16 + arow) * (unsigned)ldw + (khalf << 4);

  v16bf bf0[NT], bf1[NT], af0[MT], af1[MT];

#pragma unroll
  for (int j = 0; j < NT; j++) bf0[j] = load_bfrag(W, boff[j]);
#pragma unroll
  for (int i = 0; i < MT; i++) af0[i] = load_afrag(A, aoff[i]);

#pragma unroll 1
  for (int k = 0; k < K; k += 64) {
    // stage 0: prefetch k+32 into buf1, compute on buf0
#pragma unroll
    for (int j = 0; j < NT; j++) bf1[j] = load_bfrag(W, boff[j] + k + 32);
#pragma unroll
    for (int i = 0; i < MT; i++) af1[i] = load_afrag(A, aoff[i] + k + 32);
#pragma unroll
    for (int j = 0; j < NT; j++)
#pragma unroll
      for (int i = 0; i < MT; i++)
        acc[j][i] = __builtin_amdgcn_wmma_f32_16x16x32_bf16(
            false, af0[i], false, bf0[j], (short)0, acc[j][i], false, false);

    // stage 1: prefetch k+64 into buf0 (unless last), compute on buf1
    if (k + 64 < K) {
#pragma unroll
      for (int j = 0; j < NT; j++) bf0[j] = load_bfrag(W, boff[j] + k + 64);
#pragma unroll
      for (int i = 0; i < MT; i++) af0[i] = load_afrag(A, aoff[i] + k + 64);
    }
#pragma unroll
    for (int j = 0; j < NT; j++)
#pragma unroll
      for (int i = 0; i < MT; i++)
        acc[j][i] = __builtin_amdgcn_wmma_f32_16x16x32_bf16(
            false, af1[i], false, bf1[j], (short)0, acc[j][i], false, false);
  }
}

// ---------------------------------------------------------------------------
// Generic bf16 WMMA GEMM:  out[m,n] = relu?( A[m,:]·W[n,:] (+A2[m,:]·W2[n,:])
//                                            + bias[n] (+bias2[n]) )
// One wave owns NT 16-wide N tiles x MT 16-high M tiles.
// 256 threads = 8 waves / block.  launch_bounds(.,1) releases the full
// per-wave VGPR budget so the double-buffered fragments never spill.
// ---------------------------------------------------------------------------
template<int NT, int MT>
__global__ void __launch_bounds__(256, 1) gemm_wmma_kernel(
    const __bf16* __restrict__ A,  int lda,
    const __bf16* __restrict__ W,  int ldw, int K,
    const __bf16* __restrict__ A2, int lda2,
    const __bf16* __restrict__ W2, int ldw2, int K2,
    const float* __restrict__ bias, const float* __restrict__ bias2,
    float* __restrict__ outF, __bf16* __restrict__ outB,
    int ldo, int N, int relu)
{
  const int lane = threadIdx.x & 31;
  const int wave = threadIdx.x >> 5;
  const int n0   = (blockIdx.x * 8 + wave) * (NT * 16);
  if (n0 >= N) return;                       // wave-uniform: EXEC stays all-1s
  const int m0 = blockIdx.y * (MT * 16);

  v8f acc[NT][MT] = {};
  gemm_loop<NT, MT>(A, lda, W, ldw, K, m0, n0, lane, acc);
  if (A2) gemm_loop<NT, MT>(A2, lda2, W2, ldw2, K2, m0, n0, lane, acc);

  // Epilogue. C/D layout: VGPR r -> row (tile_m0 + r + 8*(lane>>4)), col (tile_n0 + (lane&15))
  const int rbase = m0 + ((lane >> 4) << 3);
#pragma unroll
  for (int j = 0; j < NT; j++) {
    const int col = n0 + j * 16 + (lane & 15);
    float bv = 0.f;
    if (bias)  bv += bias[col];
    if (bias2) bv += bias2[col];
    const bool colOK = (col < N);
#pragma unroll
    for (int m = 0; m < MT; m++) {
#pragma unroll
      for (int r = 0; r < 8; r++) {
        float v = acc[j][m][r] + bv;
        if (relu) v = fmaxf(v, 0.f);
        size_t off = (size_t)(rbase + m * 16 + r) * ldo + col;
        if (colOK) {
          if (outF) outF[off] = v;
          if (outB) outB[off] = (__bf16)v;
        }
      }
    }
  }
}

// ---------------------------------------------------------------------------
// Prep kernels (run once per launch; deterministic)
// ---------------------------------------------------------------------------
__global__ void cvt_bf16_kernel(const float* __restrict__ src,
                                __bf16* __restrict__ dst, int n) {
  int i = blockIdx.x * 256 + threadIdx.x;
  if (i < n) dst[i] = (__bf16)src[i];
}

// projT[n,k] = project[k,n]  (1024x1024)
__global__ void transpose_cvt_kernel(const float* __restrict__ src,
                                     __bf16* __restrict__ dst) {
  int i = blockIdx.x * 256 + threadIdx.x;   // i = n*1024 + k
  int n = i >> 10, k = i & 1023;
  dst[i] = (__bf16)src[k * 1024 + n];
}

// W_hta [1000,1024] -> zero-padded [1024,1024] bf16
__global__ void pad_whta_kernel(const float* __restrict__ w,
                                __bf16* __restrict__ dst) {
  int i = blockIdx.x * 256 + threadIdx.x;
  int r = i >> 10, c = i & 1023;
  dst[i] = (r < AS_) ? (__bf16)w[r * 1024 + c] : (__bf16)0.f;
}

__global__ void pad_bhta_kernel(const float* __restrict__ b,
                                float* __restrict__ dst) {
  int i = threadIdx.x + blockIdx.x * 256;
  if (i < ASP_) dst[i] = (i < AS_) ? b[i] : 0.f;
}

// X_bf[b,t,:] = emb[actions[b,t],:]
__global__ void gather_emb_kernel(const int* __restrict__ actions,
                                  const float* __restrict__ emb,
                                  __bf16* __restrict__ X) {
  int i = blockIdx.x * 256 + threadIdx.x;   // over B*T*E
  int e = i & (E_ - 1), bt = i >> 8;
  int a = actions[bt];
  X[i] = (__bf16)emb[a * E_ + e];
}

__global__ void init_state_kernel(const float* __restrict__ h0,
                                  const float* __restrict__ c0,
                                  float* __restrict__ h, float* __restrict__ c,
                                  __bf16* __restrict__ hbf) {
  int i = blockIdx.x * 256 + threadIdx.x;   // over B*H
  float v = h0[i];
  h[i] = v; c[i] = c0[i]; hbf[i] = (__bf16)v;
}

// c_env == current_env_context exactly (softmax over identical rows is uniform)
__global__ void fill_env_kernel(const float* __restrict__ env,
                                __bf16* __restrict__ x_cat) {
  int i = blockIdx.x * 256 + threadIdx.x;   // over B*E
  int b = i >> 8, k = i & (E_ - 1);
  x_cat[b * WD_ + (E_ + H_) + k] = (__bf16)env[i];
}

// ---------------------------------------------------------------------------
// Per-step kernels
// ---------------------------------------------------------------------------
__global__ void copyE_kernel(const __bf16* __restrict__ X,
                             __bf16* __restrict__ x_cat, int t) {
  int i = blockIdx.x * 256 + threadIdx.x;   // over B*E
  int b = i >> 8, e = i & (E_ - 1);
  x_cat[b * WD_ + e] = X[((size_t)(b * T_ + t)) * E_ + e];
}

// scores[b,s] = ins_proj[b,s,:] · h[b,:]   (one wave per score, lane-split K)
__global__ void scores_kernel(const __bf16* __restrict__ insp,
                              const float* __restrict__ h,
                              float* __restrict__ scores) {
  int gw   = (blockIdx.x * blockDim.x + threadIdx.x) >> 5;
  int lane = threadIdx.x & 31;
  int b = gw >> 7, s = gw & 127;
  const __bf16* row = insp + ((size_t)(b * S_ + s) << 10);
  const float*  hb  = h + ((size_t)b << 10);
  float sum = 0.f;
  for (int k0 = lane * 8; k0 < H_; k0 += 256) {
    v4f_t bv = *(const v4f_t*)(row + k0);        // 8 bf16
    const __bf16* bb = (const __bf16*)&bv;
    float4 ha = *(const float4*)(hb + k0);
    float4 hc = *(const float4*)(hb + k0 + 4);
    sum += (float)bb[0] * ha.x + (float)bb[1] * ha.y +
           (float)bb[2] * ha.z + (float)bb[3] * ha.w +
           (float)bb[4] * hc.x + (float)bb[5] * hc.y +
           (float)bb[6] * hc.z + (float)bb[7] * hc.w;
  }
#pragma unroll
  for (int off = 16; off; off >>= 1) sum += __shfl_xor(sum, off, 32);
  if (lane == 0) scores[gw] = sum;
}

__global__ void softmax_kernel(const float* __restrict__ scores,
                               float* __restrict__ alpha) {
  __shared__ float sm[S_];
  int b = blockIdx.x, t = threadIdx.x;
  float v = scores[b * S_ + t];
  sm[t] = v; __syncthreads();
  for (int off = 64; off; off >>= 1) {
    if (t < off) sm[t] = fmaxf(sm[t], sm[t + off]);
    __syncthreads();
  }
  float mx = sm[0]; __syncthreads();
  float e = __expf(v - mx);
  sm[t] = e; __syncthreads();
  for (int off = 64; off; off >>= 1) {
    if (t < off) sm[t] += sm[t + off];
    __syncthreads();
  }
  alpha[b * S_ + t] = e / sm[0];
}

// c_ins[b,k] = sum_s alpha[b,s] * ins[b,s,k]  -> written into x_cat slot
__global__ void context_kernel(const float* __restrict__ alpha,
                               const __bf16* __restrict__ ins_bf,
                               __bf16* __restrict__ x_cat) {
  __shared__ float al[S_];
  int b = blockIdx.x;
  int k = blockIdx.y * 256 + threadIdx.x;
  if (threadIdx.x < S_) al[threadIdx.x] = alpha[b * S_ + threadIdx.x];
  __syncthreads();
  float acc = 0.f;
  const __bf16* base = ins_bf + (((size_t)b) << 17) + k;   // b*S*H
  for (int s = 0; s < S_; s++) acc += al[s] * (float)base[(size_t)s << 10];
  x_cat[b * WD_ + E_ + k] = (__bf16)acc;
}

// LSTM pointwise: torch gate order i,f,g,o
__global__ void lstm_kernel(const float* __restrict__ gates,
                            float* __restrict__ c, float* __restrict__ h,
                            __bf16* __restrict__ h_bf,
                            __bf16* __restrict__ h_all_bf, int t) {
  int i = blockIdx.x * 256 + threadIdx.x;   // over B*H
  int b = i >> 10, j = i & 1023;
  const float* g = gates + (((size_t)b) << 12);
  float ig = g[j], fg = g[H_ + j], gg = g[2 * H_ + j], og = g[3 * H_ + j];
  float si = 1.f / (1.f + __expf(-ig));
  float sf = 1.f / (1.f + __expf(-fg));
  float so = 1.f / (1.f + __expf(-og));
  float cn = sf * c[i] + si * tanhf(gg);
  float hn = so * tanhf(cn);
  c[i] = cn; h[i] = hn; h_bf[i] = (__bf16)hn;
  h_all_bf[((size_t)(b * T_ + t)) * H_ + j] = (__bf16)hn;
}

// ---------------------------------------------------------------------------
// Launch
// ---------------------------------------------------------------------------
extern "C" void kernel_launch(void* const* d_in, const int* in_sizes, int n_in,
                              void* d_out, int out_size, void* d_ws, size_t ws_size,
                              hipStream_t stream) {
  (void)in_sizes; (void)n_in; (void)out_size; (void)ws_size;

  const float* ins     = (const float*)d_in[0];
  const int*   actions = (const int*)  d_in[2];
  const float* env     = (const float*)d_in[3];
  const float* emb     = (const float*)d_in[6];
  const float* project = (const float*)d_in[7];
  const float* W_bd    = (const float*)d_in[9];
  const float* b_bd    = (const float*)d_in[10];
  const float* W_ih    = (const float*)d_in[11];
  const float* W_hh    = (const float*)d_in[12];
  const float* b_ih    = (const float*)d_in[13];
  const float* b_hh    = (const float*)d_in[14];
  const float* W_hta   = (const float*)d_in[15];
  const float* b_hta   = (const float*)d_in[16];
  const float* h0      = (const float*)d_in[17];
  const float* c0      = (const float*)d_in[18];
  float* out = (float*)d_out;

  char* ws = (char*)d_ws;
  size_t off = 0;
  auto alloc = [&](size_t bytes) -> void* {
    void* p = ws + off;
    off += (bytes + 255) & ~(size_t)255;
    return p;
  };
  __bf16* ins_bf      = (__bf16*)alloc((size_t)B_ * S_ * H_ * 2);
  __bf16* ins_proj_bf = (__bf16*)alloc((size_t)B_ * S_ * H_ * 2);
  __bf16* projT       = (__bf16*)alloc((size_t)H_ * H_ * 2);
  __bf16* Wbd_bf      = (__bf16*)alloc((size_t)H_ * WD_ * 2);
  __bf16* Wih_bf      = (__bf16*)alloc((size_t)G_ * H_ * 2);
  __bf16* Whh_bf      = (__bf16*)alloc((size_t)G_ * H_ * 2);
  __bf16* Whta_bf     = (__bf16*)alloc((size_t)ASP_ * H_ * 2);
  float*  bhta_pad    = (float*) alloc((size_t)ASP_ * 4);
  __bf16* X_bf        = (__bf16*)alloc((size_t)B_ * T_ * E_ * 2);
  float*  h           = (float*) alloc((size_t)B_ * H_ * 4);
  float*  c           = (float*) alloc((size_t)B_ * H_ * 4);
  __bf16* h_bf        = (__bf16*)alloc((size_t)B_ * H_ * 2);
  __bf16* h_all_bf    = (__bf16*)alloc((size_t)B_ * T_ * H_ * 2);
  float*  scores      = (float*) alloc((size_t)B_ * S_ * 4);
  float*  alpha       = (float*) alloc((size_t)B_ * S_ * 4);
  __bf16* x_cat       = (__bf16*)alloc((size_t)B_ * WD_ * 2);
  __bf16* xr_bf       = (__bf16*)alloc((size_t)B_ * H_ * 2);
  float*  gates       = (float*) alloc((size_t)B_ * G_ * 4);

  dim3 blk(256);

  // ---- prep: bf16 weight/activation conversions -------------------------
  cvt_bf16_kernel<<<(G_ * H_) / 256, blk, 0, stream>>>(W_ih, Wih_bf, G_ * H_);
  cvt_bf16_kernel<<<(G_ * H_) / 256, blk, 0, stream>>>(W_hh, Whh_bf, G_ * H_);
  cvt_bf16_kernel<<<(H_ * WD_) / 256, blk, 0, stream>>>(W_bd, Wbd_bf, H_ * WD_);
  cvt_bf16_kernel<<<(B_ * S_ * H_) / 256, blk, 0, stream>>>(ins, ins_bf, B_ * S_ * H_);
  transpose_cvt_kernel<<<(H_ * H_) / 256, blk, 0, stream>>>(project, projT);
  pad_whta_kernel<<<(ASP_ * H_) / 256, blk, 0, stream>>>(W_hta, Whta_bf);
  pad_bhta_kernel<<<ASP_ / 256, blk, 0, stream>>>(b_hta, bhta_pad);
  gather_emb_kernel<<<(B_ * T_ * E_) / 256, blk, 0, stream>>>(actions, emb, X_bf);
  init_state_kernel<<<(B_ * H_) / 256, blk, 0, stream>>>(h0, c0, h, c, h_bf);
  fill_env_kernel<<<(B_ * E_) / 256, blk, 0, stream>>>(env, x_cat);

  // ---- ins_proj = ins @ project   [16384,1024]x[1024,1024] (bf16 WMMA) ---
  // NT=2,MT=4: grid.x = (1024/32)/8 = 4, grid.y = 16384/64 = 256
  gemm_wmma_kernel<2, 4><<<dim3(4, (B_ * S_) / 64), blk, 0, stream>>>(
      ins_bf, H_, projT, H_, H_,
      nullptr, 0, nullptr, 0, 0,
      nullptr, nullptr,
      nullptr, ins_proj_bf, H_, H_, 0);

  // ---- 64 sequential decoder steps --------------------------------------
  for (int t = 0; t < T_; t++) {
    scores_kernel<<<(B_ * S_) / 8, blk, 0, stream>>>(ins_proj_bf, h, scores);
    softmax_kernel<<<B_, dim3(S_), 0, stream>>>(scores, alpha);
    context_kernel<<<dim3(B_, H_ / 256), blk, 0, stream>>>(alpha, ins_bf, x_cat);
    copyE_kernel<<<(B_ * E_) / 256, blk, 0, stream>>>(X_bf, x_cat, t);

    // xr = relu(x_cat @ W_bd^T + b_bd)   -> bf16
    gemm_wmma_kernel<2, 4><<<dim3(4, B_ / 64), blk, 0, stream>>>(
        x_cat, WD_, Wbd_bf, WD_, WD_,
        nullptr, 0, nullptr, 0, 0,
        b_bd, nullptr,
        nullptr, xr_bf, H_, H_, 1);

    // gates = xr @ W_ih^T + h @ W_hh^T + b_ih + b_hh
    gemm_wmma_kernel<2, 4><<<dim3(G_ / 32 / 8, B_ / 64), blk, 0, stream>>>(
        xr_bf, H_, Wih_bf, H_, H_,
        h_bf, H_, Whh_bf, H_, H_,
        b_ih, b_hh,
        gates, nullptr, G_, G_, 0);

    lstm_kernel<<<(B_ * H_) / 256, blk, 0, stream>>>(gates, c, h, h_bf, h_all_bf, t);
  }

  // ---- logits = relu(h_all @ W_hta^T + b_hta)  [8192,1024]x[1024,1000] ---
  // tiles32 = 32 -> grid.x = 4; grid.y = 8192/64 = 128
  gemm_wmma_kernel<2, 4><<<dim3(4, (B_ * T_) / 64), blk, 0, stream>>>(
      h_all_bf, H_, Whta_bf, H_, H_,
      nullptr, 0, nullptr, 0, 0,
      bhta_pad, nullptr,
      out, nullptr, AS_, AS_, 1);
}
